// MoE_3633542332734
// MI455X (gfx1250) — matrix operand order repaired and checked by reading, not compiled, not run on hardware
//
#include <hip/hip_runtime.h>
#include <math.h>

// ---------------- problem constants ----------------
#define NTOK 4096            // B*S
#define DDIM 1024
#define FDIM 4096
#define NEXP 8
#define MAXSLOTS (2 * NTOK + 16 * NEXP)   // 8320 (top-2 slots, per-expert padded to 16)
#define MAXTILES (MAXSLOTS / 16)          // 520

typedef __attribute__((ext_vector_type(16))) __bf16 v16bf;
typedef __attribute__((ext_vector_type(8)))  __bf16 bf16x8;
typedef __attribute__((ext_vector_type(8)))  float  v8f;

// ---------------- workspace layout (bytes) ----------------
constexpr size_t OFF_CNT  = 0;                                      // int[8]
constexpr size_t OFF_CUR  = 64;                                     // int[8]
constexpr size_t OFF_POFF = 128;                                    // int[9]
constexpr size_t OFF_IDX  = 256;                                    // int[NTOK*2]
constexpr size_t OFF_PR   = OFF_IDX  + (size_t)NTOK * 2 * 4;        // float[NTOK*2]
constexpr size_t OFF_SLOT = OFF_PR   + (size_t)NTOK * 2 * 4;        // int[NTOK*2]
constexpr size_t OFF_LIST = OFF_SLOT + (size_t)NTOK * 2 * 4;        // int[MAXSLOTS]
constexpr size_t OFF_Y    = (OFF_LIST + (size_t)MAXSLOTS * 4 + 255) & ~(size_t)255; // float[MAXSLOTS*DDIM]
constexpr size_t OFF_W1S  = (OFF_Y + (size_t)MAXSLOTS * DDIM * 4 + 255) & ~(size_t)255; // bf16[E*D*F]
constexpr size_t OFF_W2S  = OFF_W1S + (size_t)NEXP * DDIM * FDIM * 2;                   // bf16[E*F*D]
// total ~ 169 MB

__device__ __forceinline__ float gelu_exact(float x) {
    return 0.5f * x * (1.0f + erff(x * 0.70710678118654752440f));
}

// ---------------- K0: init counters + pad list with token 0 ----------------
__global__ void moe_init(int* cnt, int* cur, int* list) {
    int i = blockIdx.x * blockDim.x + threadIdx.x;
    if (i < NEXP) { cnt[i] = 0; cur[i] = 0; }
    if (i < MAXSLOTS) list[i] = 0;
}

// ---------------- K1a/K1b: swizzle weights fp32 -> fragment-native bf16 ----------------
// Tile = 32(K) x 16(N). Storage: 32 lanes x 16 bf16 contiguous (1KB per tile).
// Lane L, element j: K = ktile*32 + (L>>4)*16 + j ; N = ntile*16 + (L&15)
// w1s tile order: (e, nt, kt)  -> phase-1 walks kt contiguously (1KB stride)
__global__ void moe_swizzle_w1(const float* __restrict__ w1, __bf16* __restrict__ w1s) {
    const int gid = blockIdx.x * blockDim.x + threadIdx.x;   // 2^21 threads
    const int L  = gid & 31;
    const int kt = (gid >> 5) & 31;          // D/32
    const int nt = (gid >> 10) & 255;        // F/16
    const int e  = gid >> 18;
    const int K0 = kt * 32 + (L >> 4) * 16;
    const int N  = nt * 16 + (L & 15);
    const float* src = w1 + (size_t)e * DDIM * FDIM + (size_t)K0 * FDIM + N;
    bf16x8 lo, hi;
#pragma unroll
    for (int j = 0; j < 8; j++) lo[j] = (__bf16)src[(size_t)j * FDIM];
#pragma unroll
    for (int j = 0; j < 8; j++) hi[j] = (__bf16)src[(size_t)(j + 8) * FDIM];
    bf16x8* dst = (bf16x8*)(w1s + ((((size_t)e * 256 + nt) * 32 + kt) * 512) + L * 16);
    dst[0] = lo; dst[1] = hi;
}
// w2s tile order: (e, kt, nt)  -> phase-2 walks nt contiguously (1KB stride)
__global__ void moe_swizzle_w2(const float* __restrict__ w2, __bf16* __restrict__ w2s) {
    const int gid = blockIdx.x * blockDim.x + threadIdx.x;   // 2^21 threads
    const int L  = gid & 31;
    const int nt = (gid >> 5) & 63;          // D/16
    const int kt = (gid >> 11) & 127;        // F/32
    const int e  = gid >> 18;
    const int K0 = kt * 32 + (L >> 4) * 16;
    const int N  = nt * 16 + (L & 15);
    const float* src = w2 + (size_t)e * FDIM * DDIM + (size_t)K0 * DDIM + N;
    bf16x8 lo, hi;
#pragma unroll
    for (int j = 0; j < 8; j++) lo[j] = (__bf16)src[(size_t)j * DDIM];
#pragma unroll
    for (int j = 0; j < 8; j++) hi[j] = (__bf16)src[(size_t)(j + 8) * DDIM];
    bf16x8* dst = (bf16x8*)(w2s + ((((size_t)e * 128 + kt) * 64 + nt) * 512) + L * 16);
    dst[0] = lo; dst[1] = hi;
}

// ---------------- K2: gating (one wave32 per token) ----------------
__global__ void moe_gate(const float* __restrict__ x, const float* __restrict__ gw,
                         const float* __restrict__ gb, int* __restrict__ idx,
                         float* __restrict__ pr, int* __restrict__ cnt) {
    const int wave = threadIdx.x >> 5;
    const int lane = threadIdx.x & 31;
    const int t = blockIdx.x * 8 + wave;

    float acc[NEXP];
#pragma unroll
    for (int e = 0; e < NEXP; e++) acc[e] = 0.0f;

    const float* xr = x + (size_t)t * DDIM;
    for (int d = lane; d < DDIM; d += 32) {
        float xv = xr[d];
        const float* g = gw + (size_t)d * NEXP;
#pragma unroll
        for (int e = 0; e < NEXP; e++) acc[e] += xv * g[e];
    }
#pragma unroll
    for (int e = 0; e < NEXP; e++) {
#pragma unroll
        for (int off = 16; off > 0; off >>= 1) acc[e] += __shfl_xor(acc[e], off, 32);
    }

    if (lane == 0) {
        float l[NEXP];
#pragma unroll
        for (int e = 0; e < NEXP; e++) l[e] = (acc[e] + gb[e]) / 0.9f;   // TEMP
        int b0 = 0; float v0 = l[0];
#pragma unroll
        for (int e = 1; e < NEXP; e++) if (l[e] > v0) { v0 = l[e]; b0 = e; }  // lowest-index tie-break
        int b1i = (b0 == 0) ? 1 : 0; float v1 = -3.4e38f;
#pragma unroll
        for (int e = 0; e < NEXP; e++) if (e != b0 && l[e] > v1) { v1 = l[e]; b1i = e; }
        float e1 = expf(v1 - v0);           // softmax over {v0, v1}, numerically stable
        float inv = 1.0f / (1.0f + e1);
        idx[2 * t] = b0;  idx[2 * t + 1] = b1i;
        pr[2 * t] = inv;  pr[2 * t + 1] = e1 * inv;
        atomicAdd(cnt + b0, 1);
        atomicAdd(cnt + b1i, 1);
    }
}

// ---------------- K3: padded per-expert slot offsets ----------------
__global__ void moe_offsets(const int* __restrict__ cnt, int* __restrict__ poff) {
    if (threadIdx.x == 0 && blockIdx.x == 0) {
        int o = 0;
        for (int e = 0; e < NEXP; e++) {
            poff[e] = o;
            o += ((cnt[e] + 15) >> 4) << 4;     // pad each expert region to 16 rows
        }
        poff[NEXP] = o;
    }
}

// ---------------- K4: scatter tokens to slots ----------------
__global__ void moe_scatter(const int* __restrict__ idx, const int* __restrict__ poff,
                            int* __restrict__ cur, int* __restrict__ list,
                            int* __restrict__ slot) {
    int t = blockIdx.x * blockDim.x + threadIdx.x;
    if (t >= NTOK) return;
#pragma unroll
    for (int k = 0; k < 2; k++) {
        int e = idx[2 * t + k];
        int pos = atomicAdd(cur + e, 1);
        int s = poff[e] + pos;
        list[s] = t;
        slot[2 * t + k] = s;
    }
}

// ---------------- K5: fused routed expert FFN (bf16 WMMA, f32 accum) ----------------
// One 256-thread block (8 waves) per 16-slot M-tile of one expert.
// B-fragments come from pre-swizzled bf16 weights: two aligned b128 loads, no converts.
__global__ void __launch_bounds__(256)
moe_expert_ffn(const float* __restrict__ x,
               const __bf16* __restrict__ w1s, const float* __restrict__ b1,
               const __bf16* __restrict__ w2s, const float* __restrict__ b2,
               const int* __restrict__ list, const int* __restrict__ poff,
               float* __restrict__ Yg) {
    __shared__ __bf16 Xs[16][DDIM + 16];  // rows 2080B = 16B-aligned
    __shared__ __bf16 Hs[16][144];        // rows 288B  = 16B-aligned

    const int slotBase = blockIdx.x * 16;
    const int total = poff[NEXP];
    if (slotBase >= total) return;        // block-uniform exit

    int e = NEXP - 1;
#pragma unroll
    for (int i = 0; i < NEXP; i++) { if (slotBase < poff[i + 1]) { e = i; break; } }

    const int tid = threadIdx.x;

    // ---- load & convert 16 x rows into LDS (fp32 -> bf16) ----
    {
        const int row = tid >> 4;                 // 0..15
        const int c0  = (tid & 15) * 64;          // 64 cols per thread
        const int token = list[slotBase + row];
        const float4* xp = (const float4*)(x + (size_t)token * DDIM + c0);
#pragma unroll 4
        for (int i = 0; i < 16; i++) {
            float4 v = xp[i];
            int c = c0 + i * 4;
            Xs[row][c + 0] = (__bf16)v.x;  Xs[row][c + 1] = (__bf16)v.y;
            Xs[row][c + 2] = (__bf16)v.z;  Xs[row][c + 3] = (__bf16)v.w;
        }
    }
    __syncthreads();

    const int wave = tid >> 5;
    const int lane = tid & 31;
    const int m  = lane & 15;
    const int hi = lane >> 4;

    const float* b1e = b1 + (size_t)e * FDIM;
    const float* b2e = b2 + (size_t)e * DDIM;

    v8f yacc[8];
#pragma unroll
    for (int j = 0; j < 8; j++)
#pragma unroll
        for (int r = 0; r < 8; r++) yacc[j][r] = 0.0f;

    for (int f0 = 0; f0 < FDIM; f0 += 128) {
        // ---- phase 1: this wave computes H[16, f0+wave*16 .. +16) ----
        v8f hacc;
#pragma unroll
        for (int r = 0; r < 8; r++) hacc[r] = 0.0f;
        const int n1 = f0 + wave * 16 + m;
        const int nt1 = (f0 >> 4) + wave;
        // kt-contiguous stream of 1KB tiles for this (e, nt)
        const bf16x8* bp1 = (const bf16x8*)(w1s + (((size_t)e * 256 + nt1) * 32) * 512 + lane * 16);

        for (int k0 = 0; k0 < DDIM; k0 += 32) {
            // A frag: two aligned 16B LDS reads (K = k0+hi*16..+15)
            const bf16x8* ap = (const bf16x8*)&Xs[m][k0 + hi * 16];
            bf16x8 alo = ap[0];
            bf16x8 ahi = ap[1];
            v16bf A = __builtin_shufflevector(alo, ahi, 0,1,2,3,4,5,6,7,8,9,10,11,12,13,14,15);
            // B frag: two aligned 16B global reads from swizzled weights
            bf16x8 blo = bp1[0];
            bf16x8 bhi = bp1[1];
            v16bf Bv = __builtin_shufflevector(blo, bhi, 0,1,2,3,4,5,6,7,8,9,10,11,12,13,14,15);
            __builtin_prefetch(bp1 + 128, 0, 0);       // next-next k-tile (+2KB)
            bp1 += 64;                                  // next kt tile (1KB)
            hacc = __builtin_amdgcn_wmma_f32_16x16x32_bf16(false, A, false, Bv,
                                                           (short)0, hacc, false, false);
        }
        const float b1v = b1e[n1];
#pragma unroll
        for (int r = 0; r < 8; r++) {
            float hv = gelu_exact(hacc[r] + b1v);
            Hs[r + (hi << 3)][wave * 16 + m] = (__bf16)hv;   // row M, col within chunk
        }
        __syncthreads();

        // ---- phase 2: Y[16, wave*128 .. +128) += H_chunk @ W2 ----
        for (int ks = 0; ks < 4; ks++) {
            const bf16x8* a2p = (const bf16x8*)&Hs[m][ks * 32 + hi * 16];
            bf16x8 a2lo = a2p[0];
            bf16x8 a2hi = a2p[1];
            v16bf A2 = __builtin_shufflevector(a2lo, a2hi, 0,1,2,3,4,5,6,7,8,9,10,11,12,13,14,15);
            const int kt2 = (f0 >> 5) + ks;
            const bf16x8* bp2 = (const bf16x8*)(w2s + ((((size_t)e * 128 + kt2) * 64) + wave * 8) * 512 + lane * 16);
#pragma unroll
            for (int j = 0; j < 8; j++) {
                bf16x8 blo = bp2[0];
                bf16x8 bhi = bp2[1];
                bp2 += 64;                              // next nt tile (1KB)
                v16bf B2 = __builtin_shufflevector(blo, bhi, 0,1,2,3,4,5,6,7,8,9,10,11,12,13,14,15);
                yacc[j] = __builtin_amdgcn_wmma_f32_16x16x32_bf16(false, A2, false, B2,
                                                                  (short)0, yacc[j], false, false);
            }
        }
        __syncthreads();
    }

    // ---- +b2 and store slot-indexed Y (fp32) ----
#pragma unroll
    for (int j = 0; j < 8; j++) {
        const int n2 = wave * 128 + j * 16 + m;
        const float b2v = b2e[n2];
#pragma unroll
        for (int r = 0; r < 8; r++) {
            int row = slotBase + r + (hi << 3);
            Yg[(size_t)row * DDIM + n2] = yacc[j][r] + b2v;
        }
    }
}

// ---------------- K6: weighted combine + LayerNorm ----------------
__global__ void moe_combine_ln(const float* __restrict__ Yg, const int* __restrict__ slot,
                               const float* __restrict__ pr, const float* __restrict__ ln_g,
                               const float* __restrict__ ln_b, float* __restrict__ out) {
    const int t = blockIdx.x;
    const int tid = threadIdx.x;
    const int s0 = slot[2 * t], s1 = slot[2 * t + 1];
    const float p0 = pr[2 * t], p1 = pr[2 * t + 1];
    const float* y0 = Yg + (size_t)s0 * DDIM;
    const float* y1 = Yg + (size_t)s1 * DDIM;

    float yv[4]; float sum = 0.0f, sq = 0.0f;
#pragma unroll
    for (int i = 0; i < 4; i++) {
        int d = tid + i * 256;
        float v = p0 * y0[d] + p1 * y1[d];
        yv[i] = v; sum += v; sq += v * v;
    }
    __shared__ float rs[256], rq[256];
    rs[tid] = sum; rq[tid] = sq;
    __syncthreads();
    for (int s = 128; s > 0; s >>= 1) {
        if (tid < s) { rs[tid] += rs[tid + s]; rq[tid] += rq[tid + s]; }
        __syncthreads();
    }
    const float mu  = rs[0] * (1.0f / DDIM);
    const float var = rq[0] * (1.0f / DDIM) - mu * mu;   // population variance (ddof=0)
    const float rstd = rsqrtf(var + 1e-5f);
    float* o = out + (size_t)t * DDIM;
#pragma unroll
    for (int i = 0; i < 4; i++) {
        int d = tid + i * 256;
        o[d] = (yv[i] - mu) * rstd * ln_g[d] + ln_b[d];
    }
}

// ---------------- launcher ----------------
extern "C" void kernel_launch(void* const* d_in, const int* in_sizes, int n_in,
                              void* d_out, int out_size, void* d_ws, size_t ws_size,
                              hipStream_t stream) {
    const float* x      = (const float*)d_in[0];
    const float* gate_w = (const float*)d_in[1];
    const float* gate_b = (const float*)d_in[2];
    const float* w1     = (const float*)d_in[3];
    const float* b1     = (const float*)d_in[4];
    const float* w2     = (const float*)d_in[5];
    const float* b2     = (const float*)d_in[6];
    const float* ln_g   = (const float*)d_in[7];
    const float* ln_b   = (const float*)d_in[8];
    float* out = (float*)d_out;

    char* ws = (char*)d_ws;
    int*    cnt  = (int*)(ws + OFF_CNT);
    int*    cur  = (int*)(ws + OFF_CUR);
    int*    poff = (int*)(ws + OFF_POFF);
    int*    idx  = (int*)(ws + OFF_IDX);
    float*  pr   = (float*)(ws + OFF_PR);
    int*    slot = (int*)(ws + OFF_SLOT);
    int*    list = (int*)(ws + OFF_LIST);
    float*  Yg   = (float*)(ws + OFF_Y);
    __bf16* w1s  = (__bf16*)(ws + OFF_W1S);
    __bf16* w2s  = (__bf16*)(ws + OFF_W2S);

    moe_init<<<(MAXSLOTS + 255) / 256, 256, 0, stream>>>(cnt, cur, list);
    moe_swizzle_w1<<<(NEXP * 256 * 32 * 32) / 256, 256, 0, stream>>>(w1, w1s);
    moe_swizzle_w2<<<(NEXP * 128 * 64 * 32) / 256, 256, 0, stream>>>(w2, w2s);
    moe_gate<<<NTOK / 8, 256, 0, stream>>>(x, gate_w, gate_b, idx, pr, cnt);
    moe_offsets<<<1, 32, 0, stream>>>(cnt, poff);
    moe_scatter<<<(NTOK + 255) / 256, 256, 0, stream>>>(idx, poff, cur, list, slot);
    moe_expert_ffn<<<MAXTILES, 256, 0, stream>>>(x, w1s, b1, w2s, b2, list, poff, Yg);
    moe_combine_ln<<<NTOK, 256, 0, stream>>>(Yg, slot, pr, ln_g, ln_b, out);
}
